// Attentive_Graph_PINN_22308060135552
// MI455X (gfx1250) — matrix operand order repaired and checked by reading, not compiled, not run on hardware
//
#include <hip/hip_runtime.h>

typedef __attribute__((ext_vector_type(16))) _Float16 v16h;
typedef __attribute__((ext_vector_type(8)))  float    v8f;
typedef __attribute__((ext_vector_type(4)))  unsigned int v4u;
typedef __attribute__((ext_vector_type(8)))  int      v8i;
typedef __attribute__((ext_vector_type(4)))  int      v4i;

// ---------------- problem constants ----------------
constexpr int Bt = 4096;   // batch
constexpr int Nn = 128;    // nodes

// ---------------- LDS layout (bytes) ----------------
// Region A (0..64K): adj f32 [128][128] during normalization; afterwards
// aliased by agg1 (f16 128x32), agg2 (f16 128x72), h2 (f32 128x65).
constexpr int OFF_ADJ   = 0;        // f32 [128][128] = 65536
constexpr int OFF_AGG1  = 0;        // f16 [128][32]  =  8192 (alias)
constexpr int OFF_AGG2  = 8192;     // f16 [128][72]  = 18432 (alias)
constexpr int OFF_H2    = 26624;    // f32 [128][65]  = 33280 (alias)
constexpr int OFF_SUP   = 65536;    // f16 [128][136] = 34816
constexpr int OFF_XT    = 100352;   // f16 [16][136]  =  4352   (x_str^T)
constexpr int OFF_H1T   = 104704;   // f16 [64][136]  = 17408   (h1^T)
constexpr int OFF_GC1T  = 122112;   // f16 [64][32]   =  4096   (gc1_w^T, K padded)
constexpr int OFF_GC2T  = 126208;   // f16 [64][64]   =  8192   (gc2_w^T)
constexpr int OFF_GC1B  = 134400;   // f32 [64]
constexpr int OFF_GC2B  = 134656;   // f32 [64]
constexpr int OFF_ATW   = 134912;   // f32 [64]
constexpr int OFF_NAW   = 135168;   // f32 [64]
constexpr int OFF_BNS   = 135424;   // f32 [128]
constexpr int OFF_BNB   = 135936;   // f32 [128]
constexpr int OFF_XRAW  = 136448;   // f32 [128]
constexpr int OFF_XCOV  = 136960;   // f32 [8]
constexpr int OFF_GE    = 136992;   // f32 [64]
constexpr int OFF_DINV  = 137248;   // f32 [128]
constexpr int OFF_PART  = 137760;   // f32 [256]
constexpr int OFF_SCORE = 138784;   // f32 [128]
constexpr int OFF_RED   = 139296;   // f32 [128]
constexpr int OFF_WGT   = 139808;   // f32 [128]
constexpr int OFF_SCAL  = 140320;   // f32 [4]
constexpr int SMEM_BYTES = 140352;

// ---------------- WMMA fragment loaders (wave32 layouts, CDNA5 ISA) ----
union HV { v16h v; float4 q[2]; };

// A 16x32 f16: lane L holds row (L%16); elems 0..7 = K kbase+(L/16)*8 .. +7,
// elems 8..15 = K kbase+16+(L/16)*8 .. +7.  Source: row-major f16 LDS.
__device__ inline v16h lds_a_frag(const _Float16* base, int pitch, int row,
                                  int kbase, int lane) {
  HV t;
  const _Float16* p = base + row * pitch + kbase + ((lane >> 4) << 3);
  t.q[0] = *(const float4*)(p);
  t.q[1] = *(const float4*)(p + 16);
  return t.v;
}
// B 32x16 f16: lane L holds col (L%16); elems 0..7 = K kbase+(L/16)*8 .. +7,
// elems 8..15 = K kbase+16+(L/16)*8 .. +7.
// Source: transposed (row contiguous in K) f16 LDS.
__device__ inline v16h lds_b_frag(const _Float16* baseT, int pitch, int col,
                                  int kbase, int lane) {
  HV t;
  const _Float16* p = baseT + col * pitch + kbase + ((lane >> 4) << 3);
  t.q[0] = *(const float4*)(p);
  t.q[1] = *(const float4*)(p + 16);
  return t.v;
}

__device__ inline float elu1(float x)      { return x > 0.f ? x : expm1f(x); }
__device__ inline float softplus_(float x) { return x > 20.f ? x : log1pf(__expf(x)); }

__global__ __launch_bounds__(256) void agpinn_fused(
    const float* __restrict__ x_str, const float* __restrict__ x_raw,
    const float* __restrict__ adj,   const float* __restrict__ x_cov,
    const float* __restrict__ age,
    const float* __restrict__ gc1_w, const float* __restrict__ gc1_b,
    const float* __restrict__ gc2_w, const float* __restrict__ gc2_b,
    const float* __restrict__ bn_g,  const float* __restrict__ bn_be,
    const float* __restrict__ bn_m,  const float* __restrict__ bn_v,
    const float* __restrict__ attn_w,const float* __restrict__ attn_b,
    const float* __restrict__ na_w,  const float* __restrict__ na_b,
    const float* __restrict__ bw1,   const float* __restrict__ bb1,
    const float* __restrict__ bw2,   const float* __restrict__ bb2,
    const float* __restrict__ aw1,   const float* __restrict__ ab1,
    const float* __restrict__ aw2,   const float* __restrict__ ab2,
    float* __restrict__ out)
{
  extern __shared__ char smem[];
  const int b    = blockIdx.x;
  const int t    = threadIdx.x;
  const int lane = t & 31;
  const int w    = t >> 5;          // wave id 0..7 -> owns rows 16w..16w+15

  float*    adjf = (float*)   (smem + OFF_ADJ);
  _Float16* agg1 = (_Float16*)(smem + OFF_AGG1);
  _Float16* agg2 = (_Float16*)(smem + OFF_AGG2);
  float*    h2   = (float*)   (smem + OFF_H2);
  _Float16* sup  = (_Float16*)(smem + OFF_SUP);
  _Float16* xT   = (_Float16*)(smem + OFF_XT);
  _Float16* h1T  = (_Float16*)(smem + OFF_H1T);
  _Float16* g1T  = (_Float16*)(smem + OFF_GC1T);
  _Float16* g2T  = (_Float16*)(smem + OFF_GC2T);
  float* g1b   = (float*)(smem + OFF_GC1B);
  float* g2b   = (float*)(smem + OFF_GC2B);
  float* atwS  = (float*)(smem + OFF_ATW);
  float* nawS  = (float*)(smem + OFF_NAW);
  float* bnS   = (float*)(smem + OFF_BNS);
  float* bnB   = (float*)(smem + OFF_BNB);
  float* xrawS = (float*)(smem + OFF_XRAW);
  float* xcovS = (float*)(smem + OFF_XCOV);
  float* geS   = (float*)(smem + OFF_GE);
  float* dinv  = (float*)(smem + OFF_DINV);
  float* part  = (float*)(smem + OFF_PART);
  float* scoreS= (float*)(smem + OFF_SCORE);
  float* red   = (float*)(smem + OFF_RED);
  float* wgtS  = (float*)(smem + OFF_WGT);
  float* scal  = (float*)(smem + OFF_SCAL);

  // ---------------- phase 0a: TDM -> stream adj tile (64KB) straight to LDS ---
  // 1-D tensor descriptor: 16384 f32 elements, tile == tensor, no padding.
  // Issued by wave 0 only (TDM ignores EXEC; uniform branch -> single issue),
  // tracked by TENSORcnt; the workgroup barrier below publishes it to all waves.
  if (w == 0) {
    const unsigned long long ga = (unsigned long long)(const void*)(adj + (size_t)b * Nn * Nn);
    const unsigned lds_lo = (unsigned)(size_t)(void*)adjf;   // LDS byte offset (addr[31:0])
    v4u g0;
    g0.x = 0x1u;                                             // count=1, user descriptor
    g0.y = lds_lo;                                           // lds_addr
    g0.z = (unsigned)(ga & 0xffffffffu);                     // global_addr[31:0]
    g0.w = (unsigned)((ga >> 32) & 0x1ffffffu) | (2u << 30); // global_addr[56:32] | type=2
    v8i g1;
    g1[0] = 0x00020000;                                      // data_size=2 (4 bytes)
    g1[1] = (int)((16384u & 0xffffu) << 16);                 // tensor_dim0[15:0] @ [63:48]
    g1[2] = (int)(16384u >> 16);                             // tensor_dim0[31:16] @ [79:64]
    g1[3] = (int)((16384u & 0xffffu) << 16);                 // tile_dim0=16384 @ [127:112]
    g1[4] = 0;                                               // tile_dim1=0, tile_dim2=0
    g1[5] = 16384;                                           // tensor_dim0_stride[31:0]
    g1[6] = 0;
    g1[7] = 0;
    v4i z4 = {0, 0, 0, 0};
#if defined(__clang_major__) && (__clang_major__ >= 23)
    v8i z8 = {0, 0, 0, 0, 0, 0, 0, 0};
    __builtin_amdgcn_tensor_load_to_lds(g0, g1, z4, z4, z8, 0);
#else
    __builtin_amdgcn_tensor_load_to_lds(g0, g1, z4, z4, 0);
#endif
    __builtin_amdgcn_s_wait_tensorcnt(0);
  }

  // ---------------- phase 0b: stage small operands/weights into LDS ----------
  {
    const float* xs = x_str + (size_t)b * Nn * 16;
    for (int i = t; i < 2048; i += 256) {                          // x_str^T f16
      int n = i >> 4, f = i & 15;
      xT[f * 136 + n] = (_Float16)xs[i];
    }
  }
  for (int i = t; i < 2048; i += 256) {                            // gc1_w^T, K pad->32
    int u = i >> 5, k = i & 31;
    g1T[u * 32 + k] = (k < 16) ? (_Float16)gc1_w[k * 64 + u] : (_Float16)0.f;
  }
  for (int i = t; i < 4096; i += 256) {                            // gc2_w^T
    int u = i >> 6, k = i & 63;
    g2T[u * 64 + k] = (_Float16)gc2_w[k * 64 + u];
  }
  if (t < 64) { g1b[t] = gc1_b[t]; g2b[t] = gc2_b[t]; atwS[t] = attn_w[t]; nawS[t] = na_w[t]; }
  if (t < 128) {
    float s = rsqrtf(bn_v[t] + 1e-5f) * bn_g[t];
    bnS[t] = s; bnB[t] = bn_be[t] - bn_m[t] * s;
    xrawS[t] = x_raw[(size_t)b * 128 + t];
  }
  if (t < 8) xcovS[t] = x_cov[(size_t)b * 8 + t];
  __syncthreads();

  // ---------------- phase 1: degree normalization -> support (f16) ----------
  {
    int row = t >> 1, half = t & 1;
    const float* rp = adjf + row * 128 + half * 64;
    float s = 0.f;
    for (int j = 0; j < 64; j++) s += rp[j];
    part[t] = s;
  }
  __syncthreads();
  if (t < 128) {
    float s = part[2 * t] + part[2 * t + 1] + 1.0f;                // +I diagonal
    float dv = rsqrtf(s);
    dinv[t] = (s > 0.f) ? dv : 0.f;
  }
  __syncthreads();
  for (int i = t; i < 16384; i += 256) {
    int r = i >> 7, c = i & 127;
    float v = adjf[i] + ((r == c) ? 1.f : 0.f);
    sup[r * 136 + c] = (_Float16)(dinv[r] * v * dinv[c]);
  }
  __syncthreads();   // adjf region is dead now; agg1/agg2/h2 may alias it

  // ---------------- MM1: agg1 = support(128x128) @ x(128x16) ---------------
  {
    v8f acc = {};
    const int arow = w * 16 + (lane & 15);
#pragma unroll
    for (int kt = 0; kt < 4; kt++) {
      v16h a  = lds_a_frag(sup, 136, arow, kt * 32, lane);
      v16h bf = lds_b_frag(xT, 136, (lane & 15), kt * 32, lane);
      acc = __builtin_amdgcn_wmma_f32_16x16x32_f16(false, a, false, bf, (short)0, acc, false, false);
    }
    const int col = lane & 15, rb = w * 16 + ((lane >> 4) << 3);
#pragma unroll
    for (int r = 0; r < 8; r++) agg1[(rb + r) * 32 + col] = (_Float16)acc[r];
  }
  for (int i = t; i < 2048; i += 256) {                            // zero K pad 16..31
    int r = i >> 4;
    agg1[r * 32 + 16 + (i & 15)] = (_Float16)0.f;
  }
  __syncthreads();

  // ---------------- MM2: h1 = elu(agg1(128x32) @ gc1_w(32x64) + b1) -> h1^T -
  {
    const int arow = w * 16 + (lane & 15);
    const int rb   = w * 16 + ((lane >> 4) << 3);
    v16h a = lds_a_frag(agg1, 32, arow, 0, lane);
#pragma unroll
    for (int nt = 0; nt < 4; nt++) {
      const int col = nt * 16 + (lane & 15);
      v16h bf = lds_b_frag(g1T, 32, col, 0, lane);
      v8f acc = {};
      acc = __builtin_amdgcn_wmma_f32_16x16x32_f16(false, a, false, bf, (short)0, acc, false, false);
      const float bias = g1b[col];
#pragma unroll
      for (int r = 0; r < 8; r++)
        h1T[col * 136 + rb + r] = (_Float16)elu1(acc[r] + bias);   // store transposed
    }
  }
  __syncthreads();

  // ---------------- MM3: agg2 = support(128x128) @ h1(128x64) --------------
  {
    const int arow = w * 16 + (lane & 15);
    const int rb   = w * 16 + ((lane >> 4) << 3);
    for (int nt = 0; nt < 4; nt++) {
      const int col = nt * 16 + (lane & 15);
      v8f acc = {};
#pragma unroll
      for (int kt = 0; kt < 4; kt++) {
        v16h a  = lds_a_frag(sup, 136, arow, kt * 32, lane);
        v16h bf = lds_b_frag(h1T, 136, col, kt * 32, lane);
        acc = __builtin_amdgcn_wmma_f32_16x16x32_f16(false, a, false, bf, (short)0, acc, false, false);
      }
#pragma unroll
      for (int r = 0; r < 8; r++) agg2[(rb + r) * 72 + col] = (_Float16)acc[r];
    }
  }
  __syncthreads();

  // ---------------- MM4: h2 = BN(elu(agg2 @ gc2_w + b2)) (f32) -------------
  {
    const int arow = w * 16 + (lane & 15);
    const int rb   = w * 16 + ((lane >> 4) << 3);
    for (int nt = 0; nt < 4; nt++) {
      const int col = nt * 16 + (lane & 15);
      v8f acc = {};
#pragma unroll
      for (int kt = 0; kt < 2; kt++) {
        v16h a  = lds_a_frag(agg2, 72, arow, kt * 32, lane);
        v16h bf = lds_b_frag(g2T, 64, col, kt * 32, lane);
        acc = __builtin_amdgcn_wmma_f32_16x16x32_f16(false, a, false, bf, (short)0, acc, false, false);
      }
      const float bias = g2b[col];
#pragma unroll
      for (int r = 0; r < 8; r++) {
        const int row = rb + r;
        float v = elu1(acc[r] + bias);
        h2[row * 65 + col] = v * bnS[row] + bnB[row];              // per-node BN
      }
    }
  }
  __syncthreads();

  // ---------------- attention scores / alpha_node --------------------------
  float* outPred  = out;
  float* outBeta  = out + Bt;
  float* outDec   = out + 2 * Bt;
  float* outAlpha = out + 3 * Bt;
  float* outW     = out + 3 * Bt + (size_t)Bt * Nn;

  if (t < 128) {
    float s = attn_b[0], al = na_b[0];
    const float* hr = h2 + t * 65;
    for (int c = 0; c < 64; c++) { float hv = hr[c]; s += hv * atwS[c]; al += hv * nawS[c]; }
    scoreS[t] = s;
    outAlpha[(size_t)b * 128 + t] = softplus_(al);
  }
  __syncthreads();

  // softmax over 128 nodes
  if (t < 128) red[t] = scoreS[t];
  __syncthreads();
  for (int st = 64; st > 0; st >>= 1) { if (t < st) red[t] = fmaxf(red[t], red[t + st]); __syncthreads(); }
  const float mx = red[0];
  __syncthreads();
  float ev = 0.f;
  if (t < 128) { ev = __expf(scoreS[t] - mx); red[t] = ev; }
  __syncthreads();
  for (int st = 64; st > 0; st >>= 1) { if (t < st) red[t] += red[t + st]; __syncthreads(); }
  const float ssum = red[0];
  __syncthreads();
  if (t < 128) {
    float wv = ev / ssum;
    wgtS[t] = wv;
    outW[(size_t)b * 128 + t] = wv;
  }
  __syncthreads();

  // graph embedding = sum_n h2[n,:] * w[n]   (pitch 65 -> conflict-free column walk)
  if (t < 64) {
    float g = 0.f;
    for (int n = 0; n < 128; n++) g += h2[n * 65 + t] * wgtS[n];
    geS[t] = g;
  }
  __syncthreads();

  // ---------------- beta head: sigmoid(tanh(in@W1+b1)@W2+b2) ---------------
  if (t < 64) {
    float acc = bb1[t];
    for (int k = 0; k < 200; k++) {
      float ink = (k < 128) ? xrawS[k] : (k < 136) ? xcovS[k - 128] : geS[k - 136];
      acc += ink * bw1[k * 64 + t];
    }
    red[t] = tanhf(acc) * bw2[t];
  }
  __syncthreads();
  for (int st = 32; st > 0; st >>= 1) { if (t < st) red[t] += red[t + st]; __syncthreads(); }
  if (t == 0) scal[0] = 1.f / (1.f + __expf(-(red[0] + bb2[0])));
  __syncthreads();

  // ---------------- decay head: softplus(relu(in@W1+b1)@W2+b2) -------------
  if (t < 32) {
    float acc = ab1[t];
    for (int k = 0; k < 192; k++) {
      float ink = (k < 64) ? geS[k] : xrawS[k - 64];
      acc += ink * aw1[k * 32 + t];
    }
    red[t] = fmaxf(acc, 0.f) * aw2[t];
  }
  __syncthreads();
  for (int st = 16; st > 0; st >>= 1) { if (t < st) red[t] += red[t + st]; __syncthreads(); }
  if (t == 0) {
    float dec = softplus_(red[0] + ab2[0]);
    float bet = scal[0];
    float ag  = age[b];
    outPred[b] = bet * (1.f - dec * ag);
    outBeta[b] = bet;
    outDec[b]  = dec;
  }
}

extern "C" void kernel_launch(void* const* d_in, const int* in_sizes, int n_in,
                              void* d_out, int out_size, void* d_ws, size_t ws_size,
                              hipStream_t stream) {
  (void)in_sizes; (void)n_in; (void)out_size; (void)d_ws; (void)ws_size;
  const float* p[25];
  for (int i = 0; i < 25; i++) p[i] = (const float*)d_in[i];
  (void)hipFuncSetAttribute(reinterpret_cast<const void*>(agpinn_fused),
                            hipFuncAttributeMaxDynamicSharedMemorySize, SMEM_BYTES);
  agpinn_fused<<<dim3(Bt), dim3(256), SMEM_BYTES, stream>>>(
      p[0], p[1], p[2], p[3], p[4],
      p[5], p[6], p[7], p[8],
      p[9], p[10], p[11], p[12],
      p[13], p[14], p[15], p[16],
      p[17], p[18], p[19], p[20],
      p[21], p[22], p[23], p[24],
      (float*)d_out);
}
